// SentenceToDocument_9053791060249
// MI455X (gfx1250) — matrix operand order repaired and checked by reading, not compiled, not run on hardware
//
#include <hip/hip_runtime.h>

// ---------------------------------------------------------------------------
// SentenceToDocument additive-attention pooling, single-pass, bf16 WMMA.
//   sentence_batch [256, 2048, 100] f32, user_vec [256,100], vw[50], wh[50,100],
//   wu[50,100], bw[50]  ->  out [256, 32, 100] f32
// One workgroup per (b, s) section: 64 words x 100 dims.
//   - section streamed once (nontemporal b128 loads), staged in LDS f32 + bf16
//   - projection GEMM [64x128]x[128x64] via v_wmma_f32_16x16x32_bf16
//   - fast tanh via v_exp_f32 + v_rcp_f32, softmax per 64-word section
//   - exact f32 weighted-sum pooling from the f32 LDS copy
// ---------------------------------------------------------------------------

typedef __attribute__((ext_vector_type(4)))  float          v4f;
typedef __attribute__((ext_vector_type(8)))  float          v8f;
typedef __attribute__((ext_vector_type(16))) __bf16         v16bf;
typedef __attribute__((ext_vector_type(8)))  unsigned short v8us;
typedef __attribute__((ext_vector_type(16))) unsigned short v16us;

#define UNITS      50
#define DIM        100
#define SEC        64      // words per sentence section (2048 / 32)
#define KPAD       128     // DIM padded to multiple of 32 for WMMA K
#define BF_STRIDE  136     // bf16 elems per row (272 B) -> rotates LDS banks
#define XF_STRIDE  104     // f32 elems per row

__device__ __forceinline__ unsigned short f2bf(float f) {
  unsigned int u = __builtin_bit_cast(unsigned int, f);
  u += 0x7FFFu + ((u >> 16) & 1u);          // round-to-nearest-even
  return (unsigned short)(u >> 16);
}

// tanh(x) = 1 - 2/(exp(2x)+1); exp->inf => +1, exp->0 => -1. ~5 VALU ops.
__device__ __forceinline__ float fast_tanh(float x) {
  float e = __expf(2.0f * x);                       // v_exp_f32 path
  float r = __builtin_amdgcn_rcpf(e + 1.0f);        // v_rcp_f32
  return 1.0f - 2.0f * r;
}

__device__ __forceinline__ v16bf ld_frag(const unsigned short* p0,
                                         const unsigned short* p1) {
  v8us a = *(const v8us*)p0;
  v8us b = *(const v8us*)p1;
  v16us c = __builtin_shufflevector(a, b, 0,1,2,3,4,5,6,7,8,9,10,11,12,13,14,15);
  return __builtin_bit_cast(v16bf, c);
}

__global__ __launch_bounds__(128)
void attn_pool_kernel(const float* __restrict__ sent,
                      const float* __restrict__ user,
                      const float* __restrict__ vw,
                      const float* __restrict__ wh,
                      const float* __restrict__ wu,
                      const float* __restrict__ bw,
                      float* __restrict__ out) {
  __shared__ unsigned short Xbf[SEC * BF_STRIDE];   // words, bf16, K-padded
  __shared__ unsigned short Wbf[64 * BF_STRIDE];    // wh,    bf16, N&K padded
  __shared__ float          Xf[SEC * XF_STRIDE];    // words, f32 (for pooling)
  __shared__ float          uproj[64];
  __shared__ float          vwl[64];
  __shared__ float          e_lds[64];
  __shared__ float          alpha[64];

  const int tid    = threadIdx.x;
  const int sec_id = blockIdx.x;            // b*32 + s  (sections contiguous)
  const int b      = sec_id >> 5;
  const float* src = sent + (size_t)sec_id * (SEC * DIM); // 6400 contiguous f32

  // ---- Phase 0a: stream section words into LDS (f32 + bf16) --------------
  // 6400 f32 = 1600 float4; nontemporal: streamed once, keep L2 for weights.
  {
    const v4f* src4 = (const v4f*)src;
    for (int i = tid; i < (SEC * DIM) / 4; i += 128) {
      v4f v = __builtin_nontemporal_load(&src4[i]);
      int base = i * 4;
      #pragma unroll
      for (int j = 0; j < 4; ++j) {
        int idx = base + j;
        int row = idx / DIM;
        int col = idx - row * DIM;
        Xf [row * XF_STRIDE + col] = v[j];
        Xbf[row * BF_STRIDE + col] = f2bf(v[j]);
      }
    }
  }
  // zero-pad bf16 K columns 100..127
  for (int i = tid; i < SEC * (KPAD - DIM); i += 128) {
    int row = i / (KPAD - DIM);
    int col = DIM + (i - row * (KPAD - DIM));
    Xbf[row * BF_STRIDE + col] = 0;
  }
  // ---- Phase 0b: wh -> LDS bf16 [64 x 128], padded region zeroed ---------
  // wh is re-read by all 8192 blocks: regular (RT) loads, stays L2-resident.
  for (int i = tid; i < 64 * KPAD; i += 128) {
    int row = i >> 7;
    int col = i & (KPAD - 1);
    float v = (row < UNITS && col < DIM) ? wh[row * DIM + col] : 0.0f;
    Wbf[row * BF_STRIDE + col] = f2bf(v);
  }
  // ---- Phase 0c: user projection + bias, vw (zero for padded units) ------
  if (tid < 64) {
    float s = 0.0f;
    if (tid < UNITS) {
      const v4f* wr = (const v4f*)(wu + tid * DIM);
      const v4f* ur = (const v4f*)(user + b * DIM);
      #pragma unroll 5
      for (int k = 0; k < DIM / 4; ++k) {
        v4f a = wr[k], u = ur[k];
        s = fmaf(a[0], u[0], s);
        s = fmaf(a[1], u[1], s);
        s = fmaf(a[2], u[2], s);
        s = fmaf(a[3], u[3], s);
      }
      s += bw[tid];
      vwl[tid] = vw[tid];
      uproj[tid] = s;
    } else {
      vwl[tid] = 0.0f;     // padded units contribute exactly 0 to e
      uproj[tid] = 0.0f;
    }
  }
  __syncthreads();

  // ---- Phase 1: E_pre[64,64] = X_bf16[64,128] @ wh_bf16^T via WMMA -------
  const int lane = tid & 31;
  const int wv   = tid >> 5;      // wave id -> M tile (16 words)
  const int n    = lane & 15;
  const int hi   = lane >> 4;

  v8f acc[4];
  #pragma unroll
  for (int t = 0; t < 4; ++t)
    #pragma unroll
    for (int r = 0; r < 8; ++r) acc[t][r] = 0.0f;

  const int arow = wv * 16 + n;   // A-matrix row (word) for this lane
  #pragma unroll
  for (int kt = 0; kt < 4; ++kt) {
    // A 16x32 bf16: lane holds K = hi*8 + [0..7] and 16 + hi*8 + [0..7]
    const unsigned short* ab = &Xbf[arow * BF_STRIDE + kt * 32 + hi * 8];
    v16bf afrag = ld_frag(ab, ab + 16);
    #pragma unroll
    for (int nt = 0; nt < 4; ++nt) {
      // B 32x16 bf16: lane = column N (unit), holds K = hi*8 + [0..7] and 16 + hi*8 + [0..7]
      const unsigned short* bb = &Wbf[(nt * 16 + n) * BF_STRIDE + kt * 32 + hi * 8];
      v16bf bfrag = ld_frag(bb, bb + 16);
      acc[nt] = __builtin_amdgcn_wmma_f32_16x16x32_bf16(
          false, afrag, false, bfrag, (short)0, acc[nt], false, false);
    }
  }

  // ---- Phase 2: e[word] = sum_u tanh(E_pre + uproj[u]) * vw[u] -----------
  // C/D layout: element r of lane -> (M = r + 8*hi, N = lane&15)
  float partial[8];
  #pragma unroll
  for (int r = 0; r < 8; ++r) partial[r] = 0.0f;
  #pragma unroll
  for (int nt = 0; nt < 4; ++nt) {
    int unit = nt * 16 + n;
    float up = uproj[unit];
    float vv = vwl[unit];
    #pragma unroll
    for (int r = 0; r < 8; ++r)
      partial[r] += fast_tanh(acc[nt][r] + up) * vv;
  }
  #pragma unroll
  for (int r = 0; r < 8; ++r) {
    float p = partial[r];
    p += __shfl_xor(p, 1, 32);
    p += __shfl_xor(p, 2, 32);
    p += __shfl_xor(p, 4, 32);
    p += __shfl_xor(p, 8, 32);
    partial[r] = p;
  }
  if (n == 0) {
    #pragma unroll
    for (int r = 0; r < 8; ++r)
      e_lds[wv * 16 + hi * 8 + r] = partial[r];
  }
  __syncthreads();

  // ---- Phase 3: softmax over the 64 words (one wave) ---------------------
  if (tid < 32) {
    float v0 = e_lds[tid], v1 = e_lds[tid + 32];
    float m = fmaxf(v0, v1);
    m = fmaxf(m, __shfl_xor(m, 16, 32));
    m = fmaxf(m, __shfl_xor(m,  8, 32));
    m = fmaxf(m, __shfl_xor(m,  4, 32));
    m = fmaxf(m, __shfl_xor(m,  2, 32));
    m = fmaxf(m, __shfl_xor(m,  1, 32));
    float x0 = __expf(v0 - m), x1 = __expf(v1 - m);
    float s = x0 + x1;
    s += __shfl_xor(s, 16, 32);
    s += __shfl_xor(s,  8, 32);
    s += __shfl_xor(s,  4, 32);
    s += __shfl_xor(s,  2, 32);
    s += __shfl_xor(s,  1, 32);
    float inv = 1.0f / s;
    alpha[tid]      = x0 * inv;
    alpha[tid + 32] = x1 * inv;
  }
  __syncthreads();

  // ---- Phase 4: weighted-sum pooling (f32, coalesced) --------------------
  if (tid < DIM) {
    float a = 0.0f;
    #pragma unroll 8
    for (int w = 0; w < SEC; ++w)
      a = fmaf(alpha[w], Xf[w * XF_STRIDE + tid], a);
    out[(size_t)sec_id * DIM + tid] = a;
  }
}

extern "C" void kernel_launch(void* const* d_in, const int* in_sizes, int n_in,
                              void* d_out, int out_size, void* d_ws, size_t ws_size,
                              hipStream_t stream) {
  (void)in_sizes; (void)n_in; (void)out_size; (void)d_ws; (void)ws_size;
  const float* sent = (const float*)d_in[0];
  const float* user = (const float*)d_in[1];
  const float* vw   = (const float*)d_in[2];
  const float* wh   = (const float*)d_in[3];
  const float* wu   = (const float*)d_in[4];
  const float* bw   = (const float*)d_in[5];
  float* out = (float*)d_out;

  dim3 grid(256 * 32);   // one workgroup per (b, s) section
  dim3 block(128);       // 4 waves of 32
  attn_pool_kernel<<<grid, block, 0, stream>>>(sent, user, vw, wh, wu, bw, out);
}
